// SigmoidMoE_6614249636438
// MI455X (gfx1250) — matrix operand order repaired and checked by reading, not compiled, not run on hardware
//
#include <hip/hip_runtime.h>

#define B_ 2
#define S_ 1024
#define D_ 1024
#define H_ 2048
#define E_ 8
#define K_ 2
#define T_ (B_*S_)     // 2048 tokens
#define RT_ (T_*K_)    // 4096 routed rows

typedef __attribute__((ext_vector_type(16))) __bf16 v16bf;
typedef __attribute__((ext_vector_type(8)))  __bf16 v8bf;
typedef __attribute__((ext_vector_type(8)))  float  v8f;

#if __has_builtin(__builtin_amdgcn_global_load_async_to_lds_b128) && \
    __has_builtin(__builtin_amdgcn_s_wait_asynccnt)
#define USE_ASYNC 1
typedef int v4i __attribute__((ext_vector_type(4)));
typedef __attribute__((address_space(1))) v4i as1_v4i;   // global
typedef __attribute__((address_space(3))) v4i as3_v4i;   // LDS
#else
#define USE_ASYNC 0
#endif

// Copy one 16-byte chunk global->LDS (async DMA if available).
__device__ __forceinline__ void cp16(__bf16* dst, const __bf16* src) {
#if USE_ASYNC
  __builtin_amdgcn_global_load_async_to_lds_b128(
      (as1_v4i*)src, (as3_v4i*)dst, 0, 0);
#else
  *(uint4*)dst = *(const uint4*)src;
#endif
}
__device__ __forceinline__ void cp32(__bf16* dst, const __bf16* src) {
  cp16(dst, src);
  cp16(dst + 8, src + 8);
}
__device__ __forceinline__ void zero32(__bf16* dst) {
  uint4 z = make_uint4(0u, 0u, 0u, 0u);
  ((uint4*)dst)[0] = z;
  ((uint4*)dst)[1] = z;
}
__device__ __forceinline__ void stage_fence() {
#if USE_ASYNC
  __builtin_amdgcn_s_wait_asynccnt(0);
#endif
  __syncthreads();
}

// Build a 16-element bf16 fragment from two 16-byte LDS chunks.
__device__ __forceinline__ v16bf ld_pair(const __bf16* p0, const __bf16* p1) {
  v8bf lo = *(const v8bf*)p0;
  v8bf hi = *(const v8bf*)p1;
  return __builtin_shufflevector(lo, hi, 0,1,2,3,4,5,6,7,8,9,10,11,12,13,14,15);
}

// ---------------------------------------------------------------- init
__global__ void moe_init(int* counts, int* fill) {
  int t = threadIdx.x;
  if (t < E_) { counts[t] = 0; fill[t] = 0; }
}

// ---------------------------------------------------------------- weight cvt+transpose
// src: [E][R][C] fp32  ->  dst: [E][C][R] bf16
__global__ __launch_bounds__(256) void moe_cvt_transpose(
    const float* __restrict__ src, __bf16* __restrict__ dst, int R, int C)
{
  __shared__ float tile[32][33];
  const int e  = blockIdx.z;
  const int c0 = blockIdx.x * 32;
  const int r0 = blockIdx.y * 32;
  const float* s = src + (size_t)e * R * C;
  __bf16* d = dst + (size_t)e * R * C;
  const int tx = threadIdx.x & 31;
  const int ty = threadIdx.x >> 5;   // 0..7
  #pragma unroll
  for (int i = 0; i < 32; i += 8)
    tile[ty + i][tx] = s[(size_t)(r0 + ty + i) * C + (c0 + tx)];
  __syncthreads();
  #pragma unroll
  for (int i = 0; i < 32; i += 8)
    d[(size_t)(c0 + ty + i) * R + (r0 + tx)] = (__bf16)tile[tx][ty + i];
}

// ---------------------------------------------------------------- router
__global__ __launch_bounds__(128) void moe_router(
    const float* __restrict__ x, const float* __restrict__ rw,
    const float* __restrict__ rb, __bf16* __restrict__ xbf,
    int* __restrict__ topk_idx, float* __restrict__ topk_w,
    int* __restrict__ counts, float* __restrict__ sq)
{
  __shared__ float red[E_][128];
  const int token = blockIdx.x;
  const int t = threadIdx.x;
  float acc[E_];
  #pragma unroll
  for (int e = 0; e < E_; ++e) acc[e] = 0.f;
  const float* xr = x + (size_t)token * D_;
  for (int d = t; d < D_; d += 128) {
    float xv = xr[d];
    xbf[(size_t)token * D_ + d] = (__bf16)xv;
    #pragma unroll
    for (int e = 0; e < E_; ++e) acc[e] += xv * rw[e * D_ + d];
  }
  #pragma unroll
  for (int e = 0; e < E_; ++e) red[e][t] = acc[e];
  __syncthreads();
  for (int s = 64; s > 0; s >>= 1) {
    if (t < s) {
      #pragma unroll
      for (int e = 0; e < E_; ++e) red[e][t] += red[e][t + s];
    }
    __syncthreads();
  }
  if (t == 0) {
    float sc[E_];
    float ssq = 0.f;
    #pragma unroll
    for (int e = 0; e < E_; ++e) {
      float lg = red[e][0] + rb[e];
      ssq += lg * lg;
      sc[e] = 1.f / (1.f + __expf(-lg));
    }
    sq[token] = ssq;
    int i0 = 0;
    #pragma unroll
    for (int e = 1; e < E_; ++e) if (sc[e] > sc[i0]) i0 = e;
    int i1 = (i0 == 0) ? 1 : 0;
    #pragma unroll
    for (int e = 0; e < E_; ++e) if (e != i0 && sc[e] > sc[i1]) i1 = e;
    float v0 = sc[i0], v1 = sc[i1];
    float inv = 1.f / (v0 + v1 + 1e-6f);
    topk_idx[token * 2 + 0] = i0;  topk_idx[token * 2 + 1] = i1;
    topk_w[token * 2 + 0] = v0 * inv;  topk_w[token * 2 + 1] = v1 * inv;
    atomicAdd(&counts[i0], 1);
    atomicAdd(&counts[i1], 1);
  }
}

// ---------------------------------------------------------------- scan
__global__ void moe_scan(const int* __restrict__ counts, int* __restrict__ offsets) {
  if (threadIdx.x == 0) {
    int s = 0;
    for (int e = 0; e < E_; ++e) { offsets[e] = s; s += counts[e]; }
    offsets[E_] = s;
  }
}

// ---------------------------------------------------------------- scatter
__global__ __launch_bounds__(256) void moe_scatter(
    const int* __restrict__ topk_idx, const float* __restrict__ topk_w,
    const int* __restrict__ offsets, int* __restrict__ fill,
    int* __restrict__ bucket, float* __restrict__ gate, int* __restrict__ pos)
{
  int token = blockIdx.x * 256 + threadIdx.x;
  if (token >= T_) return;
  #pragma unroll
  for (int k = 0; k < K_; ++k) {
    int e = topk_idx[token * 2 + k];
    int p = offsets[e] + atomicAdd(&fill[e], 1);
    bucket[p] = token;
    gate[p] = topk_w[token * 2 + k];
    pos[token * 2 + k] = p;
  }
}

// ---------------------------------------------------------------- GEMM1 + SwiGLU
// Block tile: 128 rows x 64 cols of H, paired with cols+H (h2 branch).
__global__ __launch_bounds__(256) void moe_gemm1(
    const __bf16* __restrict__ xbf, const __bf16* __restrict__ w12t,
    const int* __restrict__ counts, const int* __restrict__ offsets,
    const int* __restrict__ bucket, __bf16* __restrict__ hidden)
{
  __shared__ __bf16 As[128 * 32];
  __shared__ __bf16 Bs[2 * 64 * 32];   // [branch][n][k]

  const int e = blockIdx.z;
  const int cnt = counts[e];
  const int mtile = blockIdx.y;
  if (mtile * 128 >= cnt) return;
  const int base = offsets[e];
  const int n0 = blockIdx.x * 64;

  const int tid  = threadIdx.x;
  const int lane = tid & 31;
  const int wid  = tid >> 5;
  const int wm   = wid >> 2;   // 0..1 -> 64-row group
  const int wn   = wid & 3;    // 0..3 -> 16-col group

  // A/B loaders: 32B per thread each; row = tid>>1, 16-half chunk = tid&1
  const int row = tid >> 1;
  const int sel = (tid & 1) * 16;
  const int rglob = mtile * 128 + row;
  const __bf16* asrc = nullptr;
  if (rglob < cnt) asrc = xbf + (size_t)bucket[base + rglob] * D_ + sel;
  __bf16* adst = &As[row * 32 + sel];

  // B: 128 rows = 2 branches x 64 n-rows of w12t[e][branch*H + n0 + n][k]
  const int bbr = row >> 6;        // branch
  const int bnn = row & 63;        // n within tile
  const __bf16* bsrc =
      w12t + ((size_t)e * 2 * H_ + (size_t)bbr * H_ + n0 + bnn) * D_ + sel;
  __bf16* bdst = &Bs[bbr * (64 * 32) + bnn * 32 + sel];

  v8f c1[4], c2[4];
  #pragma unroll
  for (int i = 0; i < 4; ++i)
    #pragma unroll
    for (int r = 0; r < 8; ++r) { c1[i][r] = 0.f; c2[i][r] = 0.f; }

  const int crow = lane & 15;
  const int ac = (lane < 16) ? 0 : 8;   // A fragment K-offset
  const int bc = (lane < 16) ? 0 : 8;   // B fragment K-offset

  for (int k0 = 0; k0 < D_; k0 += 32) {
    if (asrc) cp32(adst, asrc + k0); else zero32(adst);
    cp32(bdst, bsrc + k0);
    if (k0 + 32 < D_) __builtin_prefetch(bsrc + k0 + 32, 0, 0);
    stage_fence();

    const __bf16* bp1 = &Bs[(wn * 16 + crow) * 32 + bc];
    const __bf16* bp2 = bp1 + 64 * 32;
    v16bf bf1 = ld_pair(bp1, bp1 + 16);
    v16bf bf2 = ld_pair(bp2, bp2 + 16);
    #pragma unroll
    for (int i = 0; i < 4; ++i) {
      const __bf16* ap = &As[(wm * 64 + i * 16 + crow) * 32 + ac];
      v16bf af = ld_pair(ap, ap + 16);
      c1[i] = __builtin_amdgcn_wmma_f32_16x16x32_bf16(false, af, false, bf1,
                                                      (short)0, c1[i], false, false);
      c2[i] = __builtin_amdgcn_wmma_f32_16x16x32_bf16(false, af, false, bf2,
                                                      (short)0, c2[i], false, false);
    }
    __syncthreads();
  }

  // epilogue: hidden = silu(h1) * h2
  const int cadd = (lane < 16) ? 0 : 8;
  const int col = n0 + wn * 16 + crow;
  #pragma unroll
  for (int i = 0; i < 4; ++i) {
    #pragma unroll
    for (int r = 0; r < 8; ++r) {
      int rl = mtile * 128 + wm * 64 + i * 16 + r + cadd;
      if (rl < cnt) {
        float h1 = c1[i][r], h2 = c2[i][r];
        float sv = h1 / (1.f + __expf(-h1));
        hidden[(size_t)(base + rl) * H_ + col] = (__bf16)(sv * h2);
      }
    }
  }
}

// ---------------------------------------------------------------- GEMM2
__global__ __launch_bounds__(256) void moe_gemm2(
    const __bf16* __restrict__ hidden, const __bf16* __restrict__ w3t,
    const int* __restrict__ counts, const int* __restrict__ offsets,
    const float* __restrict__ gate, float* __restrict__ rowout)
{
  __shared__ __bf16 As[128 * 32];
  __shared__ __bf16 Bs[64 * 32];

  const int e = blockIdx.z;
  const int cnt = counts[e];
  const int mtile = blockIdx.y;
  if (mtile * 128 >= cnt) return;
  const int base = offsets[e];
  const int n0 = blockIdx.x * 64;

  const int tid  = threadIdx.x;
  const int lane = tid & 31;
  const int wid  = tid >> 5;
  const int wm   = wid >> 2;
  const int wn   = wid & 3;

  const int row = tid >> 1;
  const int sel = (tid & 1) * 16;
  const int rglob = mtile * 128 + row;
  const __bf16* asrc = nullptr;
  if (rglob < cnt) asrc = hidden + (size_t)(base + rglob) * H_ + sel;
  __bf16* adst = &As[row * 32 + sel];

  // B: 64 n-rows of w3t[e][n0 + n][k] ; threads 0..127 stage it
  const __bf16* bsrc = nullptr;
  __bf16* bdst = nullptr;
  if (tid < 128) {
    bsrc = w3t + ((size_t)e * D_ + n0 + row) * H_ + sel;
    bdst = &Bs[row * 32 + sel];
  }

  v8f c[4];
  #pragma unroll
  for (int i = 0; i < 4; ++i)
    #pragma unroll
    for (int r = 0; r < 8; ++r) c[i][r] = 0.f;

  const int crow = lane & 15;
  const int ac = (lane < 16) ? 0 : 8;
  const int bc = (lane < 16) ? 0 : 8;

  for (int k0 = 0; k0 < H_; k0 += 32) {
    if (asrc) cp32(adst, asrc + k0); else zero32(adst);
    if (bsrc) {
      cp32(bdst, bsrc + k0);
      if (k0 + 32 < H_) __builtin_prefetch(bsrc + k0 + 32, 0, 0);
    }
    stage_fence();

    const __bf16* bp = &Bs[(wn * 16 + crow) * 32 + bc];
    v16bf bf = ld_pair(bp, bp + 16);
    #pragma unroll
    for (int i = 0; i < 4; ++i) {
      const __bf16* ap = &As[(wm * 64 + i * 16 + crow) * 32 + ac];
      v16bf af = ld_pair(ap, ap + 16);
      c[i] = __builtin_amdgcn_wmma_f32_16x16x32_bf16(false, af, false, bf,
                                                     (short)0, c[i], false, false);
    }
    __syncthreads();
  }

  const int cadd = (lane < 16) ? 0 : 8;
  const int col = n0 + wn * 16 + crow;
  #pragma unroll
  for (int i = 0; i < 4; ++i) {
    #pragma unroll
    for (int r = 0; r < 8; ++r) {
      int rl = mtile * 128 + wm * 64 + i * 16 + r + cadd;
      if (rl < cnt) {
        float g = gate[base + rl];
        rowout[(size_t)(base + rl) * D_ + col] = c[i][r] * g;
      }
    }
  }
}

// ---------------------------------------------------------------- combine (deterministic gather)
__global__ __launch_bounds__(256) void moe_combine(
    const float* __restrict__ rowout, const int* __restrict__ pos,
    float* __restrict__ out)
{
  int i = blockIdx.x * 256 + threadIdx.x;
  if (i >= T_ * D_) return;
  int t = i / D_;
  int d = i - t * D_;
  int p0 = pos[t * 2 + 0];
  int p1 = pos[t * 2 + 1];
  out[i] = rowout[(size_t)p0 * D_ + d] + rowout[(size_t)p1 * D_ + d];
}

// ---------------------------------------------------------------- aux loss (fixed-tree reduce)
__global__ __launch_bounds__(256) void moe_aux(const float* __restrict__ sq,
                                               float* __restrict__ aux)
{
  __shared__ float red[256];
  float s = 0.f;
  for (int i = threadIdx.x; i < T_; i += 256) s += sq[i];
  red[threadIdx.x] = s;
  __syncthreads();
  for (int k = 128; k > 0; k >>= 1) {
    if (threadIdx.x < k) red[threadIdx.x] += red[threadIdx.x + k];
    __syncthreads();
  }
  if (threadIdx.x == 0) *aux = 0.01f * red[0] / (float)(T_ * E_);
}

// ---------------------------------------------------------------- launch
extern "C" void kernel_launch(void* const* d_in, const int* in_sizes, int n_in,
                              void* d_out, int out_size, void* d_ws, size_t ws_size,
                              hipStream_t stream) {
  (void)in_sizes; (void)n_in; (void)out_size; (void)ws_size;
  const float* x   = (const float*)d_in[0];
  const float* rw  = (const float*)d_in[1];
  const float* rb  = (const float*)d_in[2];
  const float* w12 = (const float*)d_in[3];
  const float* w3  = (const float*)d_in[4];
  float* out = (float*)d_out;
  float* aux = out + (size_t)T_ * D_;

  char* ws = (char*)d_ws;
  __bf16* xbf    = (__bf16*)ws;  ws += (size_t)T_ * D_ * 2;          // 4 MB
  __bf16* hidden = (__bf16*)ws;  ws += (size_t)RT_ * H_ * 2;         // 16 MB
  float*  rowout = (float*)ws;   ws += (size_t)RT_ * D_ * 4;         // 16 MB
  __bf16* w12t   = (__bf16*)ws;  ws += (size_t)E_ * 2 * H_ * D_ * 2; // 67 MB
  __bf16* w3t    = (__bf16*)ws;  ws += (size_t)E_ * H_ * D_ * 2;     // 33.5 MB
  float*  sq     = (float*)ws;   ws += (size_t)T_ * 4;
  int*   topk_idx = (int*)ws;    ws += (size_t)T_ * K_ * 4;
  float* topk_w   = (float*)ws;  ws += (size_t)T_ * K_ * 4;
  int*   bucket   = (int*)ws;    ws += (size_t)RT_ * 4;
  float* gate     = (float*)ws;  ws += (size_t)RT_ * 4;
  int*   pos      = (int*)ws;    ws += (size_t)T_ * K_ * 4;
  int*   counts   = (int*)ws;    ws += E_ * 4;
  int*   fill     = (int*)ws;    ws += E_ * 4;
  int*   offsets  = (int*)ws;    ws += (E_ + 1) * 4;

  moe_init<<<1, 32, 0, stream>>>(counts, fill);
  // weights: fp32 -> bf16, transposed to n-major so GEMM staging is a byte copy
  dim3 gt1(2 * H_ / 32, D_ / 32, E_);
  moe_cvt_transpose<<<gt1, 256, 0, stream>>>(w12, w12t, D_, 2 * H_);
  dim3 gt2(D_ / 32, H_ / 32, E_);
  moe_cvt_transpose<<<gt2, 256, 0, stream>>>(w3, w3t, H_, D_);

  moe_router<<<T_, 128, 0, stream>>>(x, rw, rb, xbf, topk_idx, topk_w, counts, sq);
  moe_scan<<<1, 32, 0, stream>>>(counts, offsets);
  moe_scatter<<<T_ / 256, 256, 0, stream>>>(topk_idx, topk_w, offsets, fill,
                                            bucket, gate, pos);
  dim3 g1(H_ / 64, T_ / 128, E_);
  moe_gemm1<<<g1, 256, 0, stream>>>(xbf, w12t, counts, offsets, bucket, hidden);
  dim3 g2(D_ / 64, T_ / 128, E_);
  moe_gemm2<<<g2, 256, 0, stream>>>(hidden, w3t, counts, offsets, gate, rowout);
  moe_combine<<<(T_ * D_ + 255) / 256, 256, 0, stream>>>(rowout, pos, out);
  moe_aux<<<1, 256, 0, stream>>>(sq, aux);
}
